// SSAM_layer_55396488183926
// MI455X (gfx1250) — matrix operand order repaired and checked by reading, not compiled, not run on hardware
//
#include <hip/hip_runtime.h>
#include <hip/hip_bf16.h>

typedef __bf16 bf16_t;
typedef __attribute__((ext_vector_type(16))) __bf16 v16bf;
typedef __attribute__((ext_vector_type(8)))  __bf16 v8bf;
typedef __attribute__((ext_vector_type(4)))  __bf16 v4bf;
typedef __attribute__((ext_vector_type(8)))  float  v8f;

#define HWDIM 3136
#define CDIM  256
#define BDIM  32

// ---------------------------------------------------------------------------
// Kernel 1: dots[b][i][j] = (1/16) * sum_d X[b][d][i] * X[b][d][j]
// GEMM: A = X^T (256 x 3136), B = X (3136 x 256). 64x64 tile per block,
// 128 threads = 4 waves; double-buffered LDS. Pipeline phases are fenced with
// sched_barrier(0): [issue global loads] | [ds frag loads + WMMA] | [wait,
// cvt fp32->bf16, ds store] so next-tile load latency hides under the WMMAs.
// LDS tile layout: [channel_local][k] bf16, so A-fragments are two contiguous
// 16B runs and B-fragments one contiguous 32B run (ds_load_b128 pairs).
// ---------------------------------------------------------------------------
__global__ __launch_bounds__(128) void gram_kernel(const float* __restrict__ X,
                                                   float* __restrict__ dots)
{
    __shared__ bf16_t Ai[2][64 * 32];
    __shared__ bf16_t Aj[2][64 * 32];

    const int batch = blockIdx.z;
    const int i0 = blockIdx.x * 64;
    const int j0 = blockIdx.y * 64;
    const float* Xb = X + (size_t)batch * HWDIM * CDIM;

    const int tid  = threadIdx.x;
    const int wave = tid >> 5;
    const int lane = tid & 31;
    const int l    = lane & 15;
    const int h    = lane >> 4;

    // staging sub-tile: 4 channels x 4 k per thread
    const int ii0 = (tid & 15) * 4;   // channel within tile
    const int t0  = (tid >> 4) * 4;   // k within tile

    v8f acc[4] = {};
    float ri[4][4], rj[4][4];         // in-flight tile registers

    auto stage_load = [&](int kk) {
        #pragma unroll
        for (int t = 0; t < 4; ++t) {
            const size_t rowoff = (size_t)(kk + t0 + t) * CDIM;
            float4 vi = *(const float4*)&Xb[rowoff + i0 + ii0];
            float4 vj = *(const float4*)&Xb[rowoff + j0 + ii0];
            ri[t][0] = vi.x; ri[t][1] = vi.y; ri[t][2] = vi.z; ri[t][3] = vi.w;
            rj[t][0] = vj.x; rj[t][1] = vj.y; rj[t][2] = vj.z; rj[t][3] = vj.w;
        }
    };
    auto stage_store = [&](int buf) {
        #pragma unroll
        for (int c = 0; c < 4; ++c) {
            v4bf pi = {(bf16_t)ri[0][c], (bf16_t)ri[1][c], (bf16_t)ri[2][c], (bf16_t)ri[3][c]};
            v4bf pj = {(bf16_t)rj[0][c], (bf16_t)rj[1][c], (bf16_t)rj[2][c], (bf16_t)rj[3][c]};
            *(v4bf*)&Ai[buf][(ii0 + c) * 32 + t0] = pi;
            *(v4bf*)&Aj[buf][(ii0 + c) * 32 + t0] = pj;
        }
    };

    const int NK = HWDIM / 32;       // 98 iterations
    stage_load(0);
    stage_store(0);

    for (int it = 0; it < NK; ++it) {
        __syncthreads();
        const int cur  = it & 1;
        const bool more = (it + 1) < NK;

        // Phase 1: issue next tile's global loads (left in flight).
        if (more) stage_load((it + 1) * 32);
        __builtin_amdgcn_sched_barrier(0);

        // Phase 2: fragment loads (distinct regs -> staggered dscnt waits)
        // and the WMMA chain, executed while global loads are outstanding.
        const bf16_t* ap = &Ai[cur][(wave * 16 + l) * 32];
        v8bf alo = *(const v8bf*)(ap + 16 * h);
        v8bf ahi = *(const v8bf*)(ap + 16 * h + 8);
        v16bf afrag = __builtin_shufflevector(alo, ahi,
            0, 1, 2, 3, 4, 5, 6, 7, 8, 9, 10, 11, 12, 13, 14, 15);

        v16bf bfr[4];
        #pragma unroll
        for (int nt = 0; nt < 4; ++nt)
            bfr[nt] = *(const v16bf*)(&Aj[cur][(nt * 16 + l) * 32 + 16 * h]);
        #pragma unroll
        for (int nt = 0; nt < 4; ++nt)
            acc[nt] = __builtin_amdgcn_wmma_f32_16x16x32_bf16(
                false, afrag, false, bfr[nt], (short)0, acc[nt], false, false);
        __builtin_amdgcn_sched_barrier(0);

        // Phase 3: wait for globals, convert, store to the other LDS buffer.
        if (more) stage_store(1 - cur);
    }

    // epilogue: scale by c^-0.5 = 1/16, write fp32 dots
    float* D = dots + (size_t)batch * CDIM * CDIM;
    #pragma unroll
    for (int nt = 0; nt < 4; ++nt) {
        #pragma unroll
        for (int r = 0; r < 8; ++r) {
            int mi = i0 + wave * 16 + r + 8 * h;
            int nj = j0 + nt * 16 + l;
            D[(size_t)mi * CDIM + nj] = acc[nt][r] * 0.0625f;
        }
    }
}

// ---------------------------------------------------------------------------
// Kernel 2: row softmax over 256 elements; 1 wave per row; emit bf16 attn.
// ---------------------------------------------------------------------------
__global__ __launch_bounds__(256) void softmax_kernel(const float* __restrict__ dots,
                                                      bf16_t* __restrict__ attn)
{
    const int row  = blockIdx.x * 8 + (threadIdx.x >> 5);   // 0..8191
    const int lane = threadIdx.x & 31;

    const float4* rp = (const float4*)(dots + (size_t)row * CDIM + lane * 8);
    float4 a = rp[0], b = rp[1];
    float v[8] = {a.x, a.y, a.z, a.w, b.x, b.y, b.z, b.w};

    float m = v[0];
    #pragma unroll
    for (int i = 1; i < 8; ++i) m = fmaxf(m, v[i]);
    #pragma unroll
    for (int off = 16; off > 0; off >>= 1) m = fmaxf(m, __shfl_xor(m, off, 32));

    float s = 0.0f;
    #pragma unroll
    for (int i = 0; i < 8; ++i) { v[i] = __expf(v[i] - m); s += v[i]; }
    #pragma unroll
    for (int off = 16; off > 0; off >>= 1) s += __shfl_xor(s, off, 32);

    const float inv = 1.0f / s;
    bf16_t o[8];
    #pragma unroll
    for (int i = 0; i < 8; ++i) o[i] = (bf16_t)(v[i] * inv);
    *(v8bf*)(attn + (size_t)row * CDIM + lane * 8) = *(const v8bf*)o;
}

// ---------------------------------------------------------------------------
// Kernel 3: out[d,i] = sum_j attn[i,j] * X[d,j] + X[d,i]
// GEMM: A[d,j] = X[d,j], B[j,i] = attn[i,j]. 64(d) x 64(i) tile per block.
// A staging needs no transpose (k contiguous in global); B is a bf16 copy.
// Same fenced, double-buffered pipeline; non-temporal final stores (output is
// write-once, keep it out of L2).
// ---------------------------------------------------------------------------
__global__ __launch_bounds__(128) void apply_kernel(const float* __restrict__ X,
                                                    const bf16_t* __restrict__ attn,
                                                    float* __restrict__ out)
{
    __shared__ bf16_t Ads[2][64 * 32];  // [d_local][k=j]
    __shared__ bf16_t Bds[2][64 * 32];  // [i_local][k=j] (== attn row-major slice)

    const int batch = blockIdx.z;
    const int d0 = blockIdx.x * 64;
    const int i0 = blockIdx.y * 64;
    const float*  Xb = X    + (size_t)batch * HWDIM * CDIM;
    const bf16_t* Ab = attn + (size_t)batch * CDIM * CDIM;

    const int tid  = threadIdx.x;
    const int wave = tid >> 5;
    const int lane = tid & 31;
    const int l    = lane & 15;
    const int h    = lane >> 4;

    // staging: each thread owns 16 contiguous k of one row (half a 32-k row)
    const int srow = tid >> 1;          // tile row 0..63
    const int skol = (tid & 1) * 16;    // k offset 0 or 16

    v8f acc[4] = {};
    float ar[16];                        // in-flight A values
    uint4 br0, br1;                      // in-flight B (bf16 passthrough)

    auto stage_load = [&](int kk) {
        #pragma unroll
        for (int q = 0; q < 4; ++q) {
            float4 v = *(const float4*)&Xb[(size_t)(d0 + srow) * CDIM + kk + skol + q * 4];
            ar[q * 4 + 0] = v.x; ar[q * 4 + 1] = v.y; ar[q * 4 + 2] = v.z; ar[q * 4 + 3] = v.w;
        }
        const uint4* src = (const uint4*)&Ab[(size_t)(i0 + srow) * CDIM + kk + skol];
        br0 = src[0];
        br1 = src[1];
    };
    auto stage_store = [&](int buf) {
        v8bf p0, p1;
        #pragma unroll
        for (int q = 0; q < 8; ++q) { p0[q] = (bf16_t)ar[q]; p1[q] = (bf16_t)ar[8 + q]; }
        *(v8bf*)&Ads[buf][srow * 32 + skol]     = p0;
        *(v8bf*)&Ads[buf][srow * 32 + skol + 8] = p1;
        uint4* dst = (uint4*)&Bds[buf][srow * 32 + skol];
        dst[0] = br0;
        dst[1] = br1;
    };

    const int NK = CDIM / 32;   // 8 iterations
    stage_load(0);
    stage_store(0);

    for (int it = 0; it < NK; ++it) {
        __syncthreads();
        const int cur  = it & 1;
        const bool more = (it + 1) < NK;

        if (more) stage_load((it + 1) * 32);
        __builtin_amdgcn_sched_barrier(0);

        const bf16_t* ap = &Ads[cur][(wave * 16 + l) * 32];
        v8bf alo = *(const v8bf*)(ap + 16 * h);
        v8bf ahi = *(const v8bf*)(ap + 16 * h + 8);
        v16bf afrag = __builtin_shufflevector(alo, ahi,
            0, 1, 2, 3, 4, 5, 6, 7, 8, 9, 10, 11, 12, 13, 14, 15);

        v16bf bfr[4];
        #pragma unroll
        for (int nt = 0; nt < 4; ++nt)
            bfr[nt] = *(const v16bf*)(&Bds[cur][(nt * 16 + l) * 32 + 16 * h]);
        #pragma unroll
        for (int nt = 0; nt < 4; ++nt)
            acc[nt] = __builtin_amdgcn_wmma_f32_16x16x32_bf16(
                false, afrag, false, bfr[nt], (short)0, acc[nt], false, false);
        __builtin_amdgcn_sched_barrier(0);

        if (more) stage_store(1 - cur);
    }

    float* Ob = out + (size_t)batch * HWDIM * CDIM;
    #pragma unroll
    for (int nt = 0; nt < 4; ++nt) {
        #pragma unroll
        for (int r = 0; r < 8; ++r) {
            int d = d0 + wave * 16 + r + 8 * h;
            int i = i0 + nt * 16 + l;
            float val = acc[nt][r] + Xb[(size_t)d * CDIM + i];
            __builtin_nontemporal_store(val, &Ob[(size_t)d * CDIM + i]);
        }
    }
}

// ---------------------------------------------------------------------------
extern "C" void kernel_launch(void* const* d_in, const int* in_sizes, int n_in,
                              void* d_out, int out_size, void* d_ws, size_t ws_size,
                              hipStream_t stream)
{
    (void)in_sizes; (void)n_in; (void)out_size; (void)ws_size;
    const float* X = (const float*)d_in[0];
    float* out = (float*)d_out;

    // workspace layout: fp32 dots [32][256][256] then bf16 attn [32][256][256]
    float*  dots = (float*)d_ws;
    bf16_t* attn = (bf16_t*)((char*)d_ws + (size_t)BDIM * CDIM * CDIM * sizeof(float));

    gram_kernel   <<<dim3(4, 4, BDIM),          128, 0, stream>>>(X, dots);
    softmax_kernel<<<dim3(BDIM * CDIM / 8),     256, 0, stream>>>(dots, attn);
    apply_kernel  <<<dim3(HWDIM / 64, 4, BDIM), 128, 0, stream>>>(X, attn, out);
}
